// MambaCrossBlock_21827023798474
// MI455X (gfx1250) — matrix operand order repaired and checked by reading, not compiled, not run on hardware
//
#include <hip/hip_runtime.h>
#include <hip/hip_bf16.h>
#include <math.h>

// ---------------------------------------------------------------------------
// MambaCrossBlock for MI455X (gfx1250, wave32, WMMA f32_16x16x32_f16)
// ---------------------------------------------------------------------------
// Shapes: Bb=64, Cin=512, H=W=16 -> L=256, N=Bb*L=16384 tokens,
//         dm=256, di=384, ds=8, 2*di=768
// ---------------------------------------------------------------------------

#define Bb   64
#define CIN  512
#define LSEQ 256
#define NTOK (Bb * LSEQ)   // 16384
#define DM   256
#define DI   384
#define DS2  8
#define DIN2 768           // 2*di

typedef __attribute__((ext_vector_type(16))) _Float16 v16h;
typedef __attribute__((ext_vector_type(8)))  float    v8f;

union FragH {
    v16h     v;
    _Float16 h[16];
    uint4    q[2];
};

__device__ __forceinline__ v8f wmma_f16(v16h a, v16h b, v8f c) {
    // D = A(16x32 f16) x B(32x16 f16) + C(16x16 f32)
    return __builtin_amdgcn_wmma_f32_16x16x32_f16(
        /*neg_a=*/false, a, /*neg_b=*/false, b,
        /*c_mod=*/(short)0, c, /*reuse_a=*/false, /*reuse_b=*/false);
}

// A fragment: row-major f16 matrix [*, ldk]; lane holds row M=lane&15.
// Half e (0..7) -> K = k0 + 8*hi + e ; half e (8..15) -> K = k0 + 16 + 8*hi + (e-8)
__device__ __forceinline__ v16h load_fragA(const _Float16* rowptr, int k0, int hi) {
    FragH f;
    const _Float16* p = rowptr + k0 + 8 * hi;
    f.q[0] = *(const uint4*)(p);
    f.q[1] = *(const uint4*)(p + 16);
    return f.v;
}

// B fragment from a row-major weight W[out, ldk]; B[k,n] = W[n,k].
// Lane holds col N=lane&15 -> out channel o; half e (0..7) -> K = k0 + 8*hi + e ; half e (8..15) -> K = k0 + 16 + 8*hi + (e-8).
__device__ __forceinline__ v16h load_fragB(const _Float16* w, int o, int ldk,
                                           int k0, int hi) {
    FragH f;
    const uint4* p = (const uint4*)(w + (size_t)o * ldk + k0 + 8 * hi);
    f.q[0] = p[0];
    f.q[1] = p[2];
    return f.v;
}

__device__ __forceinline__ float clip5(float v) { return fminf(fmaxf(v, -5.f), 5.f); }
__device__ __forceinline__ float sigm(float v) { return 1.f / (1.f + expf(-v)); }

// ---------------------------------------------------------------------------
// K0a: f32 -> f16 weight conversion
// ---------------------------------------------------------------------------
__global__ void k0_cvt(const float* __restrict__ src, _Float16* __restrict__ dst, int n) {
    int t = blockIdx.x * blockDim.x + threadIdx.x;
    if (t < n) dst[t] = (_Float16)src[t];
}

// ---------------------------------------------------------------------------
// K0b: fold BN params: scale = g*rsqrt(v+eps), bias = b - m*scale
// ---------------------------------------------------------------------------
__global__ void k0_fold(const float* bnr_g, const float* bnr_b, const float* bnr_m,
                        const float* bnr_v, const float* bns_g, const float* bns_b,
                        const float* bns_m, const float* bns_v,
                        float* bnr_s, float* bnr_o, float* bns_s, float* bns_o) {
    int t = blockIdx.x * blockDim.x + threadIdx.x;
    if (t < DM) {
        float s = bnr_g[t] * rsqrtf(bnr_v[t] + 1e-5f);
        bnr_s[t] = s;
        bnr_o[t] = bnr_b[t] - bnr_m[t] * s;
    }
    if (t < CIN) {
        float s = bns_g[t] * rsqrtf(bns_v[t] + 1e-5f);
        bns_s[t] = s;
        bns_o[t] = bns_b[t] - bns_m[t] * s;
    }
}

// ---------------------------------------------------------------------------
// K0c: SSM param fold: A090 = clip(-exp(clip(Alog,-5,0)),-2,-0.01)*0.9,
//      B01 = B*0.1  (both streams)
// ---------------------------------------------------------------------------
__global__ void k0_ab(const float* AlogV, const float* Bv,
                      const float* AlogI, const float* Bi,
                      float* A090V, float* B01V, float* A090I, float* B01I) {
    int t = blockIdx.x * blockDim.x + threadIdx.x;
    if (t < DI * DS2) {
        float av = fminf(fmaxf(AlogV[t], -5.f), 0.f);
        float Av = fminf(fmaxf(-expf(av), -2.f), -0.01f);
        A090V[t] = Av * 0.9f;
        B01V[t]  = Bv[t] * 0.1f;
        float ai = fminf(fmaxf(AlogI[t], -5.f), 0.f);
        float Ai = fminf(fmaxf(-expf(ai), -2.f), -0.01f);
        A090I[t] = Ai * 0.9f;
        B01I[t]  = Bi[t] * 0.1f;
    }
}

// ---------------------------------------------------------------------------
// K1: reduce GEMM (x[b,c,l] * Wred^T) + BN + ReLU + LN1 -> xln (f16 [N, DM])
//     A tile staged through double-buffered LDS in TOKEN-MAJOR layout
//     (sA[t][k], 64B rows) so each lane's fragment is two ds_load_b128.
// ---------------------------------------------------------------------------
__global__ __launch_bounds__(256) void k1_reduce_bn_ln(
    const float* __restrict__ xV, const float* __restrict__ xI,
    const _Float16* __restrict__ WredH,
    const float* __restrict__ bnr_s, const float* __restrict__ bnr_o,
    const float* __restrict__ vg, const float* __restrict__ vb,
    const float* __restrict__ ig, const float* __restrict__ ib,
    _Float16* __restrict__ xlnV, _Float16* __restrict__ xlnI) {

    const int S = blockIdx.y;
    const float*  x   = S ? xI : xV;
    const float*  lg  = S ? ig : vg;
    const float*  lb  = S ? ib : vb;
    _Float16*     out = S ? xlnI : xlnV;

    const int tok0 = blockIdx.x * 16;
    const int b    = tok0 >> 8;
    const int l0   = tok0 & 255;
    const float* xb = x + (size_t)b * CIN * LSEQ + l0;   // xb[c*256 + t]

    __shared__ _Float16 sA[2][16 * 32];   // [buf][token][k_local], rows 64B
    __shared__ float    sT[16 * DM];      // 16 tokens x 256 channels
    __shared__ float2   sRed[256];
    __shared__ float2   sMV[16];

    const int tid  = threadIdx.x;
    const int wave = tid >> 5, lane = tid & 31;
    const int hi   = lane >> 4, l15 = lane & 15;

    const int o0 = wave * 32 + l15;
    const int o1 = o0 + 16;

    // coalesced stage: 16 consecutive tokens per channel (64B rows in global)
    auto stage = [&](int ks, int buf) {
        const int k0 = ks * 32;
        #pragma unroll
        for (int j = 0; j < 2; ++j) {
            int idx = tid + j * 256;          // 0..511
            int c = idx >> 4, t = idx & 15;
            sA[buf][t * 32 + c] = (_Float16)xb[(size_t)(k0 + c) * LSEQ + t];
        }
    };

    v8f acc0 = {}, acc1 = {};

    stage(0, 0);
    __syncthreads();

    for (int ks = 0; ks < 16; ++ks) {
        const int k0 = ks * 32;
        if (ks + 1 < 16) stage(ks + 1, (ks + 1) & 1);

        FragH fa;
        const _Float16* prow = &sA[ks & 1][l15 * 32 + 8 * hi];
        fa.q[0] = *(const uint4*)(prow);        // ds_load_b128
        fa.q[1] = *(const uint4*)(prow + 16);   // ds_load_b128

        v16h b0 = load_fragB(WredH, o0, CIN, k0, hi);
        v16h b1 = load_fragB(WredH, o1, CIN, k0, hi);
        acc0 = wmma_f16(fa.v, b0, acc0);
        acc1 = wmma_f16(fa.v, b1, acc1);
        __syncthreads();
    }

    // BN + ReLU -> LDS token-major tile
    {
        float s0 = bnr_s[o0], t0 = bnr_o[o0];
        float s1 = bnr_s[o1], t1 = bnr_o[o1];
        #pragma unroll
        for (int r = 0; r < 8; ++r) {
            int row = r + 8 * hi;
            sT[row * DM + o0] = fmaxf(acc0[r] * s0 + t0, 0.f);
            sT[row * DM + o1] = fmaxf(acc1[r] * s1 + t1, 0.f);
        }
    }
    __syncthreads();

    // LayerNorm over DM per token (16 threads per token)
    {
        int g = tid >> 4, j = tid & 15;
        float s = 0.f, ss = 0.f;
        #pragma unroll
        for (int k = 0; k < 16; ++k) {
            float v = sT[g * DM + j * 16 + k];
            s += v; ss += v * v;
        }
        sRed[tid] = make_float2(s, ss);
    }
    __syncthreads();
    if ((tid & 15) == 0) {
        int g = tid >> 4;
        float s = 0.f, ss = 0.f;
        for (int k = 0; k < 16; ++k) { float2 p = sRed[g * 16 + k]; s += p.x; ss += p.y; }
        float m = s * (1.f / DM);
        float var = ss * (1.f / DM) - m * m;
        sMV[g] = make_float2(m, rsqrtf(var + 1e-5f));
    }
    __syncthreads();
    {
        int g = tid >> 4, j = tid & 15;
        float m = sMV[g].x, rs = sMV[g].y;
        _Float16* orow = out + (size_t)(tok0 + g) * DM + j * 16;
        #pragma unroll
        for (int k = 0; k < 16; ++k) {
            int c = j * 16 + k;
            orow[k] = (_Float16)((sT[g * DM + c] - m) * rs * lg[c] + lb[c]);
        }
    }
}

// ---------------------------------------------------------------------------
// K2: in-projection GEMM xp = clip(xln @ Win^T) [N, 768]; split into
//     x_ssm (f32 [b,l,i]) and tanh(gate) (f16). Shared by std/cross paths.
// ---------------------------------------------------------------------------
__global__ __launch_bounds__(256) void k2_in_gemm(
    const _Float16* __restrict__ xlnV, const _Float16* __restrict__ xlnI,
    const _Float16* __restrict__ WinVH, const _Float16* __restrict__ WinIH,
    float* __restrict__ xssmV, float* __restrict__ xssmI,
    _Float16* __restrict__ tgV, _Float16* __restrict__ tgI) {

    const int S = blockIdx.y;
    const _Float16* A = S ? xlnI : xlnV;
    const _Float16* W = S ? WinIH : WinVH;
    float*    xs = S ? xssmI : xssmV;
    _Float16* tg = S ? tgI : tgV;

    const int tok0 = blockIdx.x * 16;
    const int tid = threadIdx.x;
    const int wave = tid >> 5, lane = tid & 31;
    const int hi = lane >> 4, l15 = lane & 15;

    const _Float16* arow = A + (size_t)(tok0 + l15) * DM;

    v8f acc[6] = {};
    for (int k0 = 0; k0 < DM; k0 += 32) {
        __builtin_prefetch(arow + k0 + 64, 0, 0);      // global_prefetch_b8
        v16h fa = load_fragA(arow, k0, hi);
        #pragma unroll
        for (int t = 0; t < 6; ++t) {
            int o = (wave + t * 8) * 16 + l15;
            v16h fb = load_fragB(W, o, DM, k0, hi);
            acc[t] = wmma_f16(fa, fb, acc[t]);
        }
    }

    #pragma unroll
    for (int t = 0; t < 6; ++t) {
        int o = (wave + t * 8) * 16 + l15;
        if (o < DI) {
            #pragma unroll
            for (int r = 0; r < 8; ++r) {
                int row = tok0 + r + 8 * hi;
                xs[(size_t)row * DI + o] = clip5(acc[t][r]);
            }
        } else {
            #pragma unroll
            for (int r = 0; r < 8; ++r) {
                int row = tok0 + r + 8 * hi;
                tg[(size_t)row * DI + (o - DI)] = (_Float16)tanhf(clip5(acc[t][r]));
            }
        }
    }
}

// ---------------------------------------------------------------------------
// K3: SSM scan, 4 variants via blockIdx.y:
//     0:(V, B_v) 1:(V, B_i) 2:(I, B_i) 3:(I, B_v).
//     One thread per (b, i); h[8] in registers; writes y*tanh(gate) as f16.
// ---------------------------------------------------------------------------
__global__ __launch_bounds__(256) void k3_scan(
    const float* __restrict__ xssmV, const float* __restrict__ xssmI,
    const _Float16* __restrict__ tgV, const _Float16* __restrict__ tgI,
    const float* __restrict__ A090V, const float* __restrict__ A090I,
    const float* __restrict__ B01V, const float* __restrict__ B01I,
    const float* __restrict__ Cv, const float* __restrict__ Ci,
    _Float16* __restrict__ yh0, _Float16* __restrict__ yh1,
    _Float16* __restrict__ yh2, _Float16* __restrict__ yh3) {

    const int k = blockIdx.y;
    const int sv = (k >> 1);                         // stream: 0,0,1,1
    const int bsel = (k == 1 || k == 2) ? 1 : 0;     // B matrix: V,I,I,V
    const float*    xsrc = sv ? xssmI : xssmV;
    const _Float16* tsrc = sv ? tgI : tgV;
    const float*    Aa = sv ? A090I : A090V;
    const float*    Cc = sv ? Ci : Cv;
    const float*    Bs = bsel ? B01I : B01V;
    _Float16* yh = (k == 0) ? yh0 : (k == 1) ? yh1 : (k == 2) ? yh2 : yh3;

    const int tg = blockIdx.x * blockDim.x + threadIdx.x;   // 0..24575
    const int i = tg % DI;
    const int b = tg / DI;

    float Ar[DS2], Br[DS2], Cr[DS2], h[DS2];
    #pragma unroll
    for (int d = 0; d < DS2; ++d) {
        Ar[d] = Aa[i * DS2 + d];
        Br[d] = Bs[i * DS2 + d];
        Cr[d] = Cc[i * DS2 + d];
        h[d] = 0.f;
    }

    const float*    xrow = xsrc + (size_t)b * LSEQ * DI + i;
    const _Float16* trow = tsrc + (size_t)b * LSEQ * DI + i;
    _Float16*       yrow = yh + (size_t)b * LSEQ * DI + i;

    for (int l = 0; l < LSEQ; ++l) {
        float xv = xrow[(size_t)l * DI];
        float y = 0.f;
        #pragma unroll
        for (int d = 0; d < DS2; ++d) {
            float hv = clip5(h[d] * Ar[d] + xv * Br[d]);
            h[d] = hv;
            y += hv * Cr[d];
        }
        y = clip5(y);
        float t = (float)trow[(size_t)l * DI];
        yrow[(size_t)l * DI] = (_Float16)(y * t);
    }
}

// ---------------------------------------------------------------------------
// K4: out-projection GEMM for both std & cross paths + clip + LN2 + blend:
//     f = a*LN2(clip(y_std@Wout^T)) + (1-a)*LN2(clip(y_x@Wout^T))  -> f16
// ---------------------------------------------------------------------------
__global__ __launch_bounds__(256) void k4_out_gemm_ln(
    const _Float16* __restrict__ yh0, const _Float16* __restrict__ yh1,
    const _Float16* __restrict__ yh2, const _Float16* __restrict__ yh3,
    const _Float16* __restrict__ WoutVH, const _Float16* __restrict__ WoutIH,
    const float* __restrict__ vg, const float* __restrict__ vb,
    const float* __restrict__ ig, const float* __restrict__ ib,
    const float* __restrict__ alpha_p,
    _Float16* __restrict__ fV, _Float16* __restrict__ fI) {

    const int S = blockIdx.y;
    const _Float16* Astd = S ? yh2 : yh0;
    const _Float16* Ax   = S ? yh3 : yh1;
    const _Float16* W    = S ? WoutIH : WoutVH;
    const float* lg = S ? ig : vg;
    const float* lb = S ? ib : vb;
    _Float16* out = S ? fI : fV;
    const float a = sigm(alpha_p[0]);

    const int tok0 = blockIdx.x * 16;
    const int tid = threadIdx.x;
    const int wave = tid >> 5, lane = tid & 31;
    const int hi = lane >> 4, l15 = lane & 15;

    __shared__ float  sT1[16 * DM];
    __shared__ float  sT2[16 * DM];
    __shared__ float4 sRed[256];
    __shared__ float4 sMV[16];

    const _Float16* arow1 = Astd + (size_t)(tok0 + l15) * DI;
    const _Float16* arow2 = Ax   + (size_t)(tok0 + l15) * DI;
    const int o0 = wave * 32 + l15;
    const int o1 = o0 + 16;

    v8f as0 = {}, as1 = {}, ax0 = {}, ax1 = {};
    for (int k0 = 0; k0 < DI; k0 += 32) {
        __builtin_prefetch(arow1 + k0 + 64, 0, 0);
        v16h fa1 = load_fragA(arow1, k0, hi);
        v16h fa2 = load_fragA(arow2, k0, hi);
        v16h fb0 = load_fragB(W, o0, DI, k0, hi);
        v16h fb1 = load_fragB(W, o1, DI, k0, hi);
        as0 = wmma_f16(fa1, fb0, as0);
        as1 = wmma_f16(fa1, fb1, as1);
        ax0 = wmma_f16(fa2, fb0, ax0);
        ax1 = wmma_f16(fa2, fb1, ax1);
    }

    #pragma unroll
    for (int r = 0; r < 8; ++r) {
        int row = r + 8 * hi;
        sT1[row * DM + o0] = clip5(as0[r]);
        sT1[row * DM + o1] = clip5(as1[r]);
        sT2[row * DM + o0] = clip5(ax0[r]);
        sT2[row * DM + o1] = clip5(ax1[r]);
    }
    __syncthreads();

    {
        int g = tid >> 4, j = tid & 15;
        float s1 = 0.f, q1 = 0.f, s2 = 0.f, q2 = 0.f;
        #pragma unroll
        for (int k = 0; k < 16; ++k) {
            float v1 = sT1[g * DM + j * 16 + k];
            float v2 = sT2[g * DM + j * 16 + k];
            s1 += v1; q1 += v1 * v1;
            s2 += v2; q2 += v2 * v2;
        }
        sRed[tid] = make_float4(s1, q1, s2, q2);
    }
    __syncthreads();
    if ((tid & 15) == 0) {
        int g = tid >> 4;
        float s1 = 0.f, q1 = 0.f, s2 = 0.f, q2 = 0.f;
        for (int k = 0; k < 16; ++k) {
            float4 p = sRed[g * 16 + k];
            s1 += p.x; q1 += p.y; s2 += p.z; q2 += p.w;
        }
        float m1 = s1 * (1.f / DM), m2 = s2 * (1.f / DM);
        float r1 = rsqrtf(q1 * (1.f / DM) - m1 * m1 + 1e-5f);
        float r2 = rsqrtf(q2 * (1.f / DM) - m2 * m2 + 1e-5f);
        sMV[g] = make_float4(m1, r1, m2, r2);
    }
    __syncthreads();
    {
        int g = tid >> 4, j = tid & 15;
        float4 mv = sMV[g];
        _Float16* orow = out + (size_t)(tok0 + g) * DM + j * 16;
        #pragma unroll
        for (int k = 0; k < 16; ++k) {
            int c = j * 16 + k;
            float l1 = (sT1[g * DM + c] - mv.x) * mv.y * lg[c] + lb[c];
            float l2 = (sT2[g * DM + c] - mv.z) * mv.w * lg[c] + lb[c];
            orow[k] = (_Float16)(a * l1 + (1.f - a) * l2);
        }
    }
}

// ---------------------------------------------------------------------------
// K5: residual 1x1 conv: out[b,co,l] = x[b,co,l] + g*(bns_s*GEMM + bns_o)
//     Output transposed through padded LDS so x reads + out stores are
//     fully coalesced against the channel-major layout.
// ---------------------------------------------------------------------------
__global__ __launch_bounds__(256) void k5_res(
    const _Float16* __restrict__ fV, const _Float16* __restrict__ fI,
    const _Float16* __restrict__ WresH,
    const float* __restrict__ bns_s, const float* __restrict__ bns_o,
    const float* __restrict__ xV, const float* __restrict__ xI,
    const float* __restrict__ gate_p,
    float* __restrict__ out) {

    const int S = blockIdx.y;
    const _Float16* A = S ? fI : fV;
    const float* xin = S ? xI : xV;
    float* op = out + (size_t)S * Bb * CIN * LSEQ;
    const float g = sigm(gate_p[0]);

    const int tok0 = blockIdx.x * 16;
    const int b = tok0 >> 8, l0 = tok0 & 255;
    const int tid = threadIdx.x;
    const int wave = tid >> 5, lane = tid & 31;
    const int hi = lane >> 4, l15 = lane & 15;

    __shared__ float sT[CIN * 17];   // [co][token], pad 17 to dodge bank conflicts

    const _Float16* arow = A + (size_t)(tok0 + l15) * DM;

    v8f acc[4] = {};
    for (int k0 = 0; k0 < DM; k0 += 32) {
        __builtin_prefetch(arow + k0 + 64, 0, 0);
        v16h fa = load_fragA(arow, k0, hi);
        #pragma unroll
        for (int t = 0; t < 4; ++t) {
            int o = (wave + t * 8) * 16 + l15;
            v16h fb = load_fragB(WresH, o, DM, k0, hi);
            acc[t] = wmma_f16(fa, fb, acc[t]);
        }
    }

    #pragma unroll
    for (int t = 0; t < 4; ++t) {
        int o = (wave + t * 8) * 16 + l15;
        float s = bns_s[o], bi = bns_o[o];
        #pragma unroll
        for (int r = 0; r < 8; ++r) {
            sT[o * 17 + (r + 8 * hi)] = s * acc[t][r] + bi;
        }
    }
    __syncthreads();

    // coalesced read-modify-write of the channel-major tensors
    const float* xrow = xin + (size_t)b * CIN * LSEQ + l0;
    float*       orow = op  + (size_t)b * CIN * LSEQ + l0;
    #pragma unroll
    for (int rep = 0; rep < 32; ++rep) {
        int idx = tid + rep * 256;            // 0..8191
        int co = idx >> 4, t = idx & 15;
        size_t gidx = (size_t)co * LSEQ + t;
        orow[gidx] = xrow[gidx] + g * sT[co * 17 + t];
    }
}

// ---------------------------------------------------------------------------
// Host launcher
// ---------------------------------------------------------------------------
extern "C" void kernel_launch(void* const* d_in, const int* in_sizes, int n_in,
                              void* d_out, int out_size, void* d_ws, size_t ws_size,
                              hipStream_t stream) {
    const float* xV    = (const float*)d_in[0];
    const float* xI    = (const float*)d_in[1];
    const float* Wred  = (const float*)d_in[2];
    const float* bnr_g = (const float*)d_in[3];
    const float* bnr_b = (const float*)d_in[4];
    const float* bnr_m = (const float*)d_in[5];
    const float* bnr_v = (const float*)d_in[6];
    const float* Wres  = (const float*)d_in[7];
    const float* bns_g = (const float*)d_in[8];
    const float* bns_b = (const float*)d_in[9];
    const float* bns_m = (const float*)d_in[10];
    const float* bns_v = (const float*)d_in[11];
    const float* v_ln1_g = (const float*)d_in[12];
    const float* v_ln1_b = (const float*)d_in[13];
    const float* v_Win   = (const float*)d_in[14];
    const float* v_Alog  = (const float*)d_in[15];
    const float* v_B     = (const float*)d_in[16];
    const float* v_C     = (const float*)d_in[17];
    const float* v_Wout  = (const float*)d_in[18];
    const float* v_ln2_g = (const float*)d_in[19];
    const float* v_ln2_b = (const float*)d_in[20];
    const float* i_ln1_g = (const float*)d_in[21];
    const float* i_ln1_b = (const float*)d_in[22];
    const float* i_Win   = (const float*)d_in[23];
    const float* i_Alog  = (const float*)d_in[24];
    const float* i_B     = (const float*)d_in[25];
    const float* i_C     = (const float*)d_in[26];
    const float* i_Wout  = (const float*)d_in[27];
    const float* i_ln2_g = (const float*)d_in[28];
    const float* i_ln2_b = (const float*)d_in[29];
    const float* alpha   = (const float*)d_in[30];
    const float* gatep   = (const float*)d_in[31];
    float* out = (float*)d_out;

    // --- workspace layout (256B aligned slabs) ---
    char* ws = (char*)d_ws;
    size_t off = 0;
    auto alloc = [&](size_t bytes) -> char* {
        char* p = ws + off;
        off += (bytes + 255) & ~(size_t)255;
        return p;
    };
    _Float16* WredH  = (_Float16*)alloc((size_t)DM * CIN * 2);
    _Float16* WinVH  = (_Float16*)alloc((size_t)DIN2 * DM * 2);
    _Float16* WinIH  = (_Float16*)alloc((size_t)DIN2 * DM * 2);
    _Float16* WoutVH = (_Float16*)alloc((size_t)DM * DI * 2);
    _Float16* WoutIH = (_Float16*)alloc((size_t)DM * DI * 2);
    _Float16* WresH  = (_Float16*)alloc((size_t)CIN * DM * 2);
    float* bnr_s = (float*)alloc(DM * 4);
    float* bnr_o = (float*)alloc(DM * 4);
    float* bns_s = (float*)alloc(CIN * 4);
    float* bns_o = (float*)alloc(CIN * 4);
    float* A090V = (float*)alloc(DI * DS2 * 4);
    float* A090I = (float*)alloc(DI * DS2 * 4);
    float* B01V  = (float*)alloc(DI * DS2 * 4);
    float* B01I  = (float*)alloc(DI * DS2 * 4);
    _Float16* xlnV = (_Float16*)alloc((size_t)NTOK * DM * 2);
    _Float16* xlnI = (_Float16*)alloc((size_t)NTOK * DM * 2);
    float* xssmV = (float*)alloc((size_t)NTOK * DI * 4);
    float* xssmI = (float*)alloc((size_t)NTOK * DI * 4);
    _Float16* tgV = (_Float16*)alloc((size_t)NTOK * DI * 2);
    _Float16* tgI = (_Float16*)alloc((size_t)NTOK * DI * 2);
    _Float16* yh0 = (_Float16*)alloc((size_t)NTOK * DI * 2);
    _Float16* yh1 = (_Float16*)alloc((size_t)NTOK * DI * 2);
    _Float16* yh2 = (_Float16*)alloc((size_t)NTOK * DI * 2);
    _Float16* yh3 = (_Float16*)alloc((size_t)NTOK * DI * 2);
    _Float16* fV  = (_Float16*)alloc((size_t)NTOK * DM * 2);
    _Float16* fI  = (_Float16*)alloc((size_t)NTOK * DM * 2);
    (void)ws_size; (void)n_in; (void)in_sizes; (void)out_size;

    // --- param prep ---
    auto cvt = [&](const float* s, _Float16* d, int n) {
        k0_cvt<<<(n + 255) / 256, 256, 0, stream>>>(s, d, n);
    };
    cvt(Wred,   WredH,  DM * CIN);
    cvt(v_Win,  WinVH,  DIN2 * DM);
    cvt(i_Win,  WinIH,  DIN2 * DM);
    cvt(v_Wout, WoutVH, DM * DI);
    cvt(i_Wout, WoutIH, DM * DI);
    cvt(Wres,   WresH,  CIN * DM);
    k0_fold<<<2, 256, 0, stream>>>(bnr_g, bnr_b, bnr_m, bnr_v,
                                   bns_g, bns_b, bns_m, bns_v,
                                   bnr_s, bnr_o, bns_s, bns_o);
    k0_ab<<<(DI * DS2 + 255) / 256, 256, 0, stream>>>(
        v_Alog, v_B, i_Alog, i_B, A090V, B01V, A090I, B01I);

    // --- pipeline ---
    dim3 blk(256);
    k1_reduce_bn_ln<<<dim3(NTOK / 16, 2), blk, 0, stream>>>(
        xV, xI, WredH, bnr_s, bnr_o,
        v_ln1_g, v_ln1_b, i_ln1_g, i_ln1_b, xlnV, xlnI);

    k2_in_gemm<<<dim3(NTOK / 16, 2), blk, 0, stream>>>(
        xlnV, xlnI, WinVH, WinIH, xssmV, xssmI, tgV, tgI);

    k3_scan<<<dim3((Bb * DI) / 256, 4), blk, 0, stream>>>(
        xssmV, xssmI, tgV, tgI, A090V, A090I, B01V, B01I,
        v_C, i_C, yh0, yh1, yh2, yh3);

    k4_out_gemm_ln<<<dim3(NTOK / 16, 2), blk, 0, stream>>>(
        yh0, yh1, yh2, yh3, WoutVH, WoutIH,
        v_ln2_g, v_ln2_b, i_ln2_g, i_ln2_b, alpha, fV, fI);

    k5_res<<<dim3(NTOK / 16, 2), blk, 0, stream>>>(
        fV, fI, WresH, bns_s, bns_o, xV, xI, gatep, out);
}
